// LeakyRNNModel_40450001993764
// MI455X (gfx1250) — matrix operand order repaired and checked by reading, not compiled, not run on hardware
//
#include <hip/hip_runtime.h>
#include <stdint.h>

// ---------------------------------------------------------------------------
// LeakyRNN on MI455X (gfx1250, wave32): bf16 WMMA recurrent steps with
// async global->LDS staging of W_hh (ASYNCcnt path), WMMA output GEMM,
// device-side fixed-point warmup control (graph-capturable).
// ---------------------------------------------------------------------------

typedef __bf16 bf16_t;
typedef __attribute__((ext_vector_type(16))) __bf16 v16bf;
typedef __attribute__((ext_vector_type(8)))  __bf16 v8bf;
typedef __attribute__((ext_vector_type(8)))  float  v8f;

constexpr int T_STEPS = 256;
constexpr int BATCH   = 64;
constexpr int NIN     = 256;
constexpr int HID     = 2048;
constexpr int NOUT    = 256;
constexpr float DECAY       = 0.2f;               // DT/TAU = 20/100
constexpr float NOISE_SCALE = 0.15811388300842f;  // 0.05*sqrt(2/0.2)
constexpr float ATOL        = 0.1f;

constexpr int KCHUNK = 256;                       // K tile staged in LDS
constexpr int BROWS  = 32;                        // 2 N-tiles worth of W_hh rows

struct Ctl { unsigned done, stable, copyflag, maxbits; };

// ----------------------------- RNG (stateless) -----------------------------
__device__ __forceinline__ unsigned mix32(unsigned x) {
  x ^= x >> 17; x *= 0xed5ad4bbu;
  x ^= x >> 11; x *= 0xac4c1b51u;
  x ^= x >> 15; x *= 0x31848babu;
  x ^= x >> 14;
  return x;
}
__device__ __forceinline__ float gauss_noise(unsigned phase, unsigned step,
                                             unsigned row, unsigned col) {
  unsigned s  = mix32(phase * 0x9e3779b9u ^ mix32(step + 0x85ebca6bu));
  unsigned u1 = mix32(s ^ (row * 0x01000193u + col * 0x0001b3u + 0x12345u));
  unsigned u2 = mix32(u1 ^ 0xdeadbeefu);
  float f1 = ((float)u1 + 0.5f) * 2.3283064365386963e-10f;   // (0,1)
  float f2 = (float)(u2 >> 8) * 5.9604644775390625e-08f;     // [0,1)
  float r  = __fsqrt_rn(-2.0f * __logf(f1));
  return r * __cosf(6.28318530717958647f * f2);
}

// ------------------- async global->LDS copy (CDNA5 path) -------------------
// Same INST_OFFSET is added to both the global and LDS side (ISA 08 §4.4),
// so one address pair serves 4 consecutive B128 copies.
__device__ __forceinline__ void async_copy4_b128(unsigned lds_addr,
                                                 unsigned long long gaddr) {
  asm volatile("global_load_async_to_lds_b128 %0, %1, off"
               :: "v"(lds_addr), "v"(gaddr) : "memory");
  asm volatile("global_load_async_to_lds_b128 %0, %1, off offset:16"
               :: "v"(lds_addr), "v"(gaddr) : "memory");
  asm volatile("global_load_async_to_lds_b128 %0, %1, off offset:32"
               :: "v"(lds_addr), "v"(gaddr) : "memory");
  asm volatile("global_load_async_to_lds_b128 %0, %1, off offset:48"
               :: "v"(lds_addr), "v"(gaddr) : "memory");
}
__device__ __forceinline__ void wait_asynccnt0() {
  asm volatile("s_wait_asynccnt 0x0" ::: "memory");
}

// ----------------------------- small utilities -----------------------------
__global__ void f2bf_kernel(const float* __restrict__ s, bf16_t* __restrict__ d, int n) {
  for (int i = blockIdx.x * blockDim.x + threadIdx.x; i < n; i += gridDim.x * blockDim.x)
    d[i] = (bf16_t)s[i];
}

__global__ void zero_state_kernel(float* __restrict__ hf, bf16_t* __restrict__ hb, int n) {
  for (int i = blockIdx.x * blockDim.x + threadIdx.x; i < n; i += gridDim.x * blockDim.x) {
    hf[i] = 0.0f;
    hb[i] = (bf16_t)0.0f;
  }
}

__global__ void ctl_init_kernel(Ctl* ctl) {
  ctl->done = 0u; ctl->stable = 0u; ctl->copyflag = 0u; ctl->maxbits = 0u;
}

__global__ __launch_bounds__(256)
void maxdiff_kernel(const float* __restrict__ a, const float* __restrict__ b,
                    Ctl* ctl, int n) {
  if (ctl->done) return;
  float m = 0.0f;
  for (int i = blockIdx.x * blockDim.x + threadIdx.x; i < n; i += gridDim.x * blockDim.x)
    m = fmaxf(m, fabsf(a[i] - b[i]));
  __shared__ float sm[256];
  sm[threadIdx.x] = m;
  __syncthreads();
  for (int s = 128; s > 0; s >>= 1) {
    if ((int)threadIdx.x < s) sm[threadIdx.x] = fmaxf(sm[threadIdx.x], sm[threadIdx.x + s]);
    __syncthreads();
  }
  if (threadIdx.x == 0) atomicMax(&ctl->maxbits, __float_as_uint(sm[0]));
}

__global__ void init_update_kernel(Ctl* ctl) {
  if (ctl->done) { ctl->copyflag = 0u; return; }
  float mx = __uint_as_float(ctl->maxbits);
  bool close = (mx <= ATOL);
  unsigned stable = close ? (ctl->stable + 1u) : 0u;
  bool trigger = close && (stable >= 4u);
  ctl->stable   = stable;
  ctl->done     = trigger ? 1u : 0u;
  ctl->copyflag = trigger ? 0u : 1u;
  ctl->maxbits  = 0u;
}

__global__ void cond_copy_kernel(const float* __restrict__ src,
                                 float* __restrict__ df, bf16_t* __restrict__ db,
                                 const Ctl* ctl, int n) {
  if (!ctl->copyflag) return;
  for (int i = blockIdx.x * blockDim.x + threadIdx.x; i < n; i += gridDim.x * blockDim.x) {
    float v = src[i];
    df[i] = v;
    db[i] = (bf16_t)v;
  }
}

// ----------------------------- WMMA helpers --------------------------------
// A fragment, 16x32 bf16 (ISA 7.12.2): lane L holds row M=L%16;
// elements 0..7 -> K = 8*half + e, elements 8..15 -> K = 16 + 8*half + e.
__device__ __forceinline__ v16bf load_a_bf16(const bf16_t* __restrict__ rowp,
                                             int kbase, int half) {
  v8bf lo = *(const v8bf*)(rowp + kbase + 8 * half);
  v8bf hi = *(const v8bf*)(rowp + kbase + 16 + 8 * half);
  v16bf a;
#pragma unroll
  for (int i = 0; i < 8; ++i) { a[i] = lo[i]; a[i + 8] = hi[i]; }
  return a;
}
// B fragment, 32x16 bf16: lane L holds col N=L%16; K = 8*half + e (e<8), 16 + 8*half + (e-8) (e>=8).
__device__ __forceinline__ v16bf load_b_bf16(const bf16_t* colp, int kbase, int half) {
  v8bf lo = *(const v8bf*)(colp + kbase + 8 * half);
  v8bf hi = *(const v8bf*)(colp + kbase + 16 + 8 * half);
  v16bf b;
#pragma unroll
  for (int i = 0; i < 8; ++i) { b[i] = lo[i]; b[i + 8] = hi[i]; }
  return b;
}

// ----------------------------- RNN step kernel -----------------------------
// pre = h_in @ Whh^T + x_t @ Wih^T + b + noise ; h_out = 0.8 h_in + 0.2 relu(pre)
// Grid: 64 blocks x 256 thr (8 waves). Block -> 2 N-tiles x 4 M-tiles.
// W_hh rows for the block's 2 N-tiles are async-staged into LDS per K-chunk
// and shared by all 8 waves (4x less L2 traffic on the 8 MB weight matrix).
__global__ __launch_bounds__(256)
void rnn_step_kernel(const bf16_t* __restrict__ h_in_bf,
                     const float*  __restrict__ h_in_f,
                     const bf16_t* __restrict__ x_bf, int x_row_stride,
                     const bf16_t* __restrict__ Whh,   // [HID][HID] row-major (B = Whh^T)
                     const bf16_t* __restrict__ Wih,   // [HID][NIN] row-major
                     const float*  __restrict__ bias,
                     float*  __restrict__ h_out_f,
                     bf16_t* __restrict__ h_out_bf,
                     float*  __restrict__ hidden_out,  // nullable
                     const Ctl* __restrict__ ctl, int guarded,
                     unsigned noise_phase, unsigned noise_step) {
  if (guarded && ctl->done) return;   // uniform scalar branch

  __shared__ bf16_t smemB[BROWS * KCHUNK];   // 16 KB: 32 W_hh rows x 256 K

  const int tid   = threadIdx.x;
  const int lane  = tid & 31;
  const int wave  = tid >> 5;
  const int mtile = wave & 3;                 // 4 M tiles (batch)
  const int ntl   = wave >> 2;                // 0/1: local N tile
  const int ntile = blockIdx.x * 2 + ntl;     // 128 N tiles (hidden)
  const int half  = lane >> 4;
  const int l16   = lane & 15;

  // async-copy assignment: thread -> (row r of 32, 64-byte column segment)
  const int cp_r  = tid >> 3;                 // 32 rows, 8 threads each
  const int cp_c  = (tid & 7) * 32;           // elements (64 bytes)
  const unsigned lds_base =
      (unsigned)(uintptr_t)smemB + (unsigned)(cp_r * KCHUNK + cp_c) * 2u;
  const bf16_t* gsrc_row = Whh + (size_t)(blockIdx.x * BROWS + cp_r) * HID + cp_c;

  const bf16_t* arow  = h_in_bf + (size_t)(mtile * 16 + l16) * HID;
  const bf16_t* bsrow = smemB + (size_t)(ntl * 16 + l16) * KCHUNK;

  v8f acc = {};
  // K over hidden state (h @ Whh^T): 8 chunks x 8 bf16 WMMAs
  for (int kc = 0; kc < HID; kc += KCHUNK) {
    async_copy4_b128(lds_base, (unsigned long long)(gsrc_row + kc));
    wait_asynccnt0();
    __syncthreads();
#pragma unroll
    for (int kb = 0; kb < KCHUNK; kb += 32) {
      v16bf a = load_a_bf16(arow, kc + kb, half);
      v16bf b = load_b_bf16(bsrow, kb, half);
      acc = __builtin_amdgcn_wmma_f32_16x16x32_bf16(false, a, false, b,
                                                    (short)0, acc, false, false);
    }
    __syncthreads();
  }
  // K over input (x_t @ Wih^T), 8 bf16 WMMAs; x_row_stride==0 => broadcast b=0
  const bf16_t* xrow = x_bf + (size_t)(mtile * 16 + l16) * x_row_stride;
  const bf16_t* wrow = Wih + (size_t)(ntile * 16 + l16) * NIN;
#pragma unroll
  for (int kb = 0; kb < NIN; kb += 32) {
    v16bf a = load_a_bf16(xrow, kb, half);
    v16bf b = load_b_bf16(wrow, kb, half);
    acc = __builtin_amdgcn_wmma_f32_16x16x32_bf16(false, a, false, b,
                                                  (short)0, acc, false, false);
  }

  // Epilogue on C layout: row m = v + 8*half, col n = l16.
#pragma unroll
  for (int v = 0; v < 8; ++v) {
    const int m = mtile * 16 + v + 8 * half;
    const int n = ntile * 16 + l16;
    float pre  = acc[v] + bias[n] +
                 NOISE_SCALE * gauss_noise(noise_phase, noise_step, (unsigned)m, (unsigned)n);
    float relu = pre > 0.0f ? pre : 0.0f;
    float hnew = (1.0f - DECAY) * h_in_f[(size_t)m * HID + n] + DECAY * relu;
    h_out_f [(size_t)m * HID + n] = hnew;
    h_out_bf[(size_t)m * HID + n] = (bf16_t)hnew;
    if (hidden_out) hidden_out[(size_t)m * HID + n] = hnew;
  }
}

// ----------------------------- output GEMM ---------------------------------
// out[T*B, NOUT] = hidden[T*B, HID] @ Wout[NOUT, HID]^T  (A: f32->bf16 in-reg)
__global__ __launch_bounds__(256)
void out_gemm_kernel(const float* __restrict__ hidden,
                     const bf16_t* __restrict__ Wout,
                     float* __restrict__ out) {
  const int lane  = threadIdx.x & 31;
  const int wave  = threadIdx.x >> 5;
  const int tile  = blockIdx.x * 8 + wave;   // 16384 tiles = 1024 (M) * 16 (N)
  const int ntile = tile & 15;
  const int mtile = tile >> 4;
  const int half  = lane >> 4;
  const int l16   = lane & 15;

  const float*  arow = hidden + (size_t)(mtile * 16 + l16) * HID;
  const bf16_t* brow = Wout + (size_t)(ntile * 16 + l16) * HID;

  v8f acc = {};
#pragma unroll 4
  for (int kb = 0; kb < HID; kb += 32) {
    v8f flo = *(const v8f*)(arow + kb + 8 * half);
    v8f fhi = *(const v8f*)(arow + kb + 16 + 8 * half);
    v16bf a;
#pragma unroll
    for (int i = 0; i < 8; ++i) { a[i] = (bf16_t)flo[i]; a[i + 8] = (bf16_t)fhi[i]; }
    v16bf b = load_b_bf16(brow, kb, half);
    acc = __builtin_amdgcn_wmma_f32_16x16x32_bf16(false, a, false, b,
                                                  (short)0, acc, false, false);
  }
#pragma unroll
  for (int v = 0; v < 8; ++v)
    out[(size_t)(mtile * 16 + v + 8 * half) * NOUT + ntile * 16 + l16] = acc[v];
}

// ----------------------------- host driver ---------------------------------
extern "C" void kernel_launch(void* const* d_in, const int* in_sizes, int n_in,
                              void* d_out, int out_size, void* d_ws, size_t ws_size,
                              hipStream_t stream) {
  const float* x    = (const float*)d_in[0];  // [T][B][NIN]
  const float* Wih  = (const float*)d_in[1];  // [HID][NIN]
  const float* Whh  = (const float*)d_in[2];  // [HID][HID]
  const float* bias = (const float*)d_in[3];  // [HID]
  const float* Wout = (const float*)d_in[4];  // [NOUT][HID]
  (void)in_sizes; (void)n_in; (void)out_size; (void)ws_size;

  float* out_output = (float*)d_out;                                   // T*B*NOUT
  float* out_hidden = out_output + (size_t)T_STEPS * BATCH * NOUT;     // T*B*HID

  char* ws = (char*)d_ws;
  auto carve = [&](size_t bytes) -> char* {
    char* p = ws; ws += (bytes + 255) & ~(size_t)255; return p;
  };
  Ctl*    ctl    = (Ctl*)   carve(sizeof(Ctl));
  float*  hA_f   = (float*) carve((size_t)BATCH * HID * 4);
  float*  hB_f   = (float*) carve((size_t)BATCH * HID * 4);
  float*  hC_f   = (float*) carve((size_t)BATCH * HID * 4);
  bf16_t* hA_b   = (bf16_t*)carve((size_t)BATCH * HID * 2);
  bf16_t* hB_b   = (bf16_t*)carve((size_t)BATCH * HID * 2);
  bf16_t* hC_b   = (bf16_t*)carve((size_t)BATCH * HID * 2);
  bf16_t* x_b    = (bf16_t*)carve((size_t)T_STEPS * BATCH * NIN * 2);
  bf16_t* Wih_b  = (bf16_t*)carve((size_t)HID * NIN * 2);
  bf16_t* Whh_b  = (bf16_t*)carve((size_t)HID * HID * 2);
  bf16_t* Wout_b = (bf16_t*)carve((size_t)NOUT * HID * 2);

  // 1) precision staging
  f2bf_kernel<<<4096, 256, 0, stream>>>(x,    x_b,    T_STEPS * BATCH * NIN);
  f2bf_kernel<<<1024, 256, 0, stream>>>(Wih,  Wih_b,  HID * NIN);
  f2bf_kernel<<<4096, 256, 0, stream>>>(Whh,  Whh_b,  HID * HID);
  f2bf_kernel<<<1024, 256, 0, stream>>>(Wout, Wout_b, NOUT * HID);
  zero_state_kernel<<<256, 256, 0, stream>>>(hA_f, hA_b, BATCH * HID);
  ctl_init_kernel<<<1, 1, 0, stream>>>(ctl);

  // 2) fixed-point warmup: h1=cell(h0,x[0,0]), h2=cell(h1,x[1,0]), converge on h2~h0
  for (int i = 0; i < 100; ++i) {
    rnn_step_kernel<<<64, 256, 0, stream>>>(hA_b, hA_f, x_b, 0,
        Whh_b, Wih_b, bias, hB_f, hB_b, nullptr, ctl, 1, 0u, (unsigned)i);
    rnn_step_kernel<<<64, 256, 0, stream>>>(hB_b, hB_f, x_b + (size_t)BATCH * NIN, 0,
        Whh_b, Wih_b, bias, hC_f, hC_b, nullptr, ctl, 1, 1u, (unsigned)i);
    maxdiff_kernel<<<128, 256, 0, stream>>>(hC_f, hA_f, ctl, BATCH * HID);
    init_update_kernel<<<1, 1, 0, stream>>>(ctl);
    cond_copy_kernel<<<128, 256, 0, stream>>>(hC_f, hA_f, hA_b, ctl, BATCH * HID);
  }

  // 3) sequential scan over T (ping-pong hA <-> hB), hidden streamed to d_out
  for (int t = 0; t < T_STEPS; ++t) {
    const bf16_t* inB = (t & 1) ? hB_b : hA_b;
    const float*  inF = (t & 1) ? hB_f : hA_f;
    bf16_t* oB = (t & 1) ? hA_b : hB_b;
    float*  oF = (t & 1) ? hA_f : hB_f;
    rnn_step_kernel<<<64, 256, 0, stream>>>(inB, inF,
        x_b + (size_t)t * BATCH * NIN, NIN,
        Whh_b, Wih_b, bias, oF, oB,
        out_hidden + (size_t)t * BATCH * HID, ctl, 0, 2u, (unsigned)t);
  }

  // 4) output projection
  out_gemm_kernel<<<2048, 256, 0, stream>>>(out_hidden, Wout_b, out_output);
}
